// DirectedChainSDE_79637283603161
// MI455X (gfx1250) — matrix operand order repaired and checked by reading, not compiled, not run on hardware
//
#include <hip/hip_runtime.h>
#include <hip/hip_bf16.h>

// ---------------------------------------------------------------------------
// DirectedChainSDE on MI455X (gfx1250).
// 255 sequential SDE steps; each step = two 4-layer MLPs as bf16 WMMA GEMMs.
// Each workgroup owns 32 batch rows for the full trajectory (rows are
// independent), state + activations in LDS, weights streamed from L2 as
// pre-packed WMMA fragments. f32 accumulate via v_wmma_f32_16x16x32_bf16.
// ---------------------------------------------------------------------------

typedef __attribute__((ext_vector_type(16))) __bf16 bf16x16;
typedef __attribute__((ext_vector_type(8)))  float  v8f;

#define HIDDEN 64
#define NOISE  32
#define MLPW   512
#define BATCH  4096
#define LEN    256
#define CH     65              // 1 + HIDDEN channels of nei / out
#define ROWSTR (LEN * CH)      // per-batch-row stride (floats)
#define INSZ   129             // 1 + 2*HIDDEN
#define KP0    160             // INSZ padded to multiple of 32
#define BM     32              // batch rows per workgroup
#define BLOCK  256             // 8 waves
#define NRT    (BM / 16)       // 2 row tiles
#define NCG    (8 / NRT)       // 4 column groups of waves

struct SdeParams {
  const float*  ts;
  const float*  nei;
  const float*  x0;
  const float*  bm;
  const __bf16* w[8];          // packed bf16 fragment weights (ws)
  const float*  b[8];          // fp32 biases (direct from d_in)
  float*        out;
};

// B fragment: 16 contiguous bf16 per lane in packed buffer.
__device__ __forceinline__ bf16x16 load_bfrag(const __bf16* p) {
  union { bf16x16 v; int4 q[2]; } u;
  u.q[0] = *(const int4*)(p);
  u.q[1] = *(const int4*)(p + 8);
  return u.v;
}
// A fragment from LDS row-major activations, per ISA 16-bit A layout:
// lane<16 -> K {0..7, 16..23}; lane>=16 -> K {8..15, 24..31} (p pre-offset
// by khalf*8; second quad sits +16 elements later).
__device__ __forceinline__ bf16x16 load_afrag(const __bf16* p) {
  union { bf16x16 v; int4 q[2]; } u;
  u.q[0] = *(const int4*)(p);
  u.q[1] = *(const int4*)(p + 16);
  return u.v;
}

__device__ __forceinline__ float lipswish(float x) {
  return 0.909f * x / (1.0f + __expf(-x));   // 0.909 * silu(x)
}

// C[BM x (CT*NCG*16)] = act(A @ W + b), A bf16 in LDS, result bf16 to LDS.
template<int KT, int CT, bool ACT>
__device__ __forceinline__ void gemm_act(const __bf16* __restrict__ in, int ldin,
                                         __bf16* __restrict__ outb, int ldout,
                                         const __bf16* __restrict__ wp,
                                         const float* __restrict__ bias) {
  const int lane = threadIdx.x & 31;
  const int wid  = threadIdx.x >> 5;
  const int rt   = wid & (NRT - 1);
  const int cg   = wid / NRT;
  const int col  = lane & 15;
  const int kh   = lane >> 4;
  const int NTt  = CT * NCG;
  v8f acc[CT];
  #pragma unroll
  for (int c = 0; c < CT; ++c)
    #pragma unroll
    for (int v = 0; v < 8; ++v) acc[c][v] = 0.0f;
  const __bf16* arow = in + (rt * 16 + col) * ldin + kh * 8;
  #pragma unroll
  for (int kt = 0; kt < KT; ++kt) {
    bf16x16 a = load_afrag(arow + kt * 32);
    const __bf16* wk = wp + (size_t)(kt * NTt) * 512 + lane * 16;
    #pragma unroll
    for (int c = 0; c < CT; ++c) {
      bf16x16 bfr = load_bfrag(wk + (size_t)(cg * CT + c) * 512);
      acc[c] = __builtin_amdgcn_wmma_f32_16x16x32_bf16(
          false, a, false, bfr, (short)0, acc[c], false, false);
    }
  }
  #pragma unroll
  for (int c = 0; c < CT; ++c) {
    const int nt = cg * CT + c;
    const float bc = bias[nt * 16 + col];
    #pragma unroll
    for (int v = 0; v < 8; ++v) {
      float x = acc[c][v] + bc;
      if (ACT) x = lipswish(x);
      const int m = rt * 16 + kh * 8 + v;
      outb[m * ldout + nt * 16 + col] = (__bf16)x;
    }
  }
}

// Drift head (512 -> 64, no activation): xnew = x + (acc + b) * dt.
// Tiles are disjoint across waves, no atomics needed.
template<int KT>
__device__ __forceinline__ void gemm_drift(const __bf16* __restrict__ in, int ldin,
                                           const __bf16* __restrict__ wp,
                                           const float* __restrict__ bias,
                                           const float* __restrict__ xb,
                                           float* __restrict__ xn, float dt) {
  const int lane = threadIdx.x & 31;
  const int wid  = threadIdx.x >> 5;
  const int rt   = wid & (NRT - 1);
  const int cg   = wid / NRT;
  const int col  = lane & 15;
  const int kh   = lane >> 4;
  v8f acc;
  #pragma unroll
  for (int v = 0; v < 8; ++v) acc[v] = 0.0f;
  const __bf16* arow = in + (rt * 16 + col) * ldin + kh * 8;
  #pragma unroll
  for (int kt = 0; kt < KT; ++kt) {
    bf16x16 a = load_afrag(arow + kt * 32);
    bf16x16 bfr = load_bfrag(wp + (size_t)(kt * 4 + cg) * 512 + lane * 16);
    acc = __builtin_amdgcn_wmma_f32_16x16x32_bf16(
        false, a, false, bfr, (short)0, acc, false, false);
  }
  const int gcol = cg * 16 + col;
  const float bc = bias[gcol];
  #pragma unroll
  for (int v = 0; v < 8; ++v) {
    const int m = rt * 16 + kh * 8 + v;
    xn[m * HIDDEN + gcol] = xb[m * HIDDEN + gcol] + (acc[v] + bc) * dt;
  }
}

// Diffusion head chunk (512 -> one 512-col slice of 2048): contract the
// accumulators against dW immediately (einsum 'bhn,bn->bh') into xnew.
template<int KT, int CT>
__device__ __forceinline__ void gemm_diff(const __bf16* __restrict__ in, int ldin,
                                          const __bf16* __restrict__ wp,
                                          const float* __restrict__ bias,
                                          int ntBase, int NTtot,
                                          const float* __restrict__ dWl,
                                          float* __restrict__ xn) {
  const int lane = threadIdx.x & 31;
  const int wid  = threadIdx.x >> 5;
  const int rt   = wid & (NRT - 1);
  const int cg   = wid / NRT;
  const int col  = lane & 15;
  const int kh   = lane >> 4;
  v8f acc[CT];
  #pragma unroll
  for (int c = 0; c < CT; ++c)
    #pragma unroll
    for (int v = 0; v < 8; ++v) acc[c][v] = 0.0f;
  const __bf16* arow = in + (rt * 16 + col) * ldin + kh * 8;
  #pragma unroll
  for (int kt = 0; kt < KT; ++kt) {
    bf16x16 a = load_afrag(arow + kt * 32);
    const __bf16* wk = wp + ((size_t)kt * NTtot + ntBase) * 512 + lane * 16;
    #pragma unroll
    for (int c = 0; c < CT; ++c) {
      bf16x16 bfr = load_bfrag(wk + (size_t)(cg * CT + c) * 512);
      acc[c] = __builtin_amdgcn_wmma_f32_16x16x32_bf16(
          false, a, false, bfr, (short)0, acc[c], false, false);
    }
  }
  #pragma unroll
  for (int c = 0; c < CT; ++c) {
    const int ntg  = ntBase + cg * CT + c;
    const int gcol = ntg * 16 + col;        // flat index h*32 + n
    const float bc = bias[gcol];
    const int h = gcol >> 5;
    const int n = gcol & 31;
    #pragma unroll
    for (int v = 0; v < 8; ++v) {
      const int m = rt * 16 + kh * 8 + v;
      atomicAdd(&xn[m * HIDDEN + h], (acc[c][v] + bc) * dWl[m * NOISE + n]);
    }
  }
}

__global__ __launch_bounds__(BLOCK)
void sde_main_kernel(SdeParams P) {
  extern __shared__ char smem[];
  __bf16* zb  = (__bf16*)smem;                 // [BM][KP0]   bf16  (10 KB)
  __bf16* hA  = zb + BM * KP0;                 // [BM][512]   bf16  (32 KB)
  __bf16* hB  = hA + BM * MLPW;                // [BM][512]   bf16  (32 KB)
  float*  xb  = (float*)(hB + BM * MLPW);      // [BM][64]    f32   (8 KB)
  float*  xn  = xb + BM * HIDDEN;              // [BM][64]    f32   (8 KB)
  float*  dWl = xn + BM * HIDDEN;              // [BM][32]    f32   (4 KB)

  const int tid = threadIdx.x;
  const int b0  = blockIdx.x * BM;

  // t = 0: x = initial; out[:,0,:] = [ts[0], x]
  for (int idx = tid; idx < BM * HIDDEN; idx += BLOCK) {
    const int r = idx / HIDDEN, c = idx - r * HIDDEN;
    const float v = P.x0[(size_t)(b0 + r) * HIDDEN + c];
    xb[idx] = v;
    P.out[(size_t)(b0 + r) * ROWSTR + 1 + c] = v;
  }
  const float t0v = P.ts[0];
  for (int r = tid; r < BM; r += BLOCK)
    P.out[(size_t)(b0 + r) * ROWSTR] = t0v;
  __syncthreads();

  for (int i = 0; i < LEN - 1; ++i) {
    const float dt  = P.ts[i + 1] - P.ts[i];
    const float sdt = sqrtf(dt);

    // Build z = [x | nei_i] as bf16 in LDS, zero-padded 129 -> 160.
    for (int idx = tid; idx < BM * KP0; idx += BLOCK) {
      const int r = idx / KP0, c = idx - r * KP0;
      float v = 0.0f;
      if (c < HIDDEN) {
        v = xb[r * HIDDEN + c];
      } else if (c < INSZ) {
        const float* np =
            &P.nei[(size_t)(b0 + r) * ROWSTR + (size_t)i * CH + (c - HIDDEN)];
        v = *np;
        __builtin_prefetch(np + CH, 0, 1);   // next timestep's nei row
      }
      zb[idx] = (__bf16)v;
    }
    // dW = bm_i * sqrt(dt)
    for (int idx = tid; idx < BM * NOISE; idx += BLOCK) {
      const int r = idx / NOISE, n = idx - r * NOISE;
      dWl[idx] = P.bm[(size_t)i * BATCH * NOISE + (size_t)(b0 + r) * NOISE + n] * sdt;
    }
    __syncthreads();

    // Drift MLP: z -> hA -> hB -> hA -> xnew (init with x + drift*dt)
    gemm_act<5, 8, true>(zb, KP0, hA, MLPW, P.w[0], P.b[0]); __syncthreads();
    gemm_act<16, 8, true>(hA, MLPW, hB, MLPW, P.w[1], P.b[1]); __syncthreads();
    gemm_act<16, 8, true>(hB, MLPW, hA, MLPW, P.w[2], P.b[2]); __syncthreads();
    gemm_drift<16>(hA, MLPW, P.w[3], P.b[3], xb, xn, dt);      __syncthreads();

    // Diffusion MLP: z -> hB -> hA -> hB, head contracted against dW.
    gemm_act<5, 8, true>(zb, KP0, hB, MLPW, P.w[4], P.b[4]); __syncthreads();
    gemm_act<16, 8, true>(hB, MLPW, hA, MLPW, P.w[5], P.b[5]); __syncthreads();
    gemm_act<16, 8, true>(hA, MLPW, hB, MLPW, P.w[6], P.b[6]); __syncthreads();
    for (int chk = 0; chk < 4; ++chk)        // 2048 cols in 4 x 512 slices
      gemm_diff<16, 8>(hB, MLPW, P.w[7], P.b[7], chk * 32, 128, dWl, xn);
    __syncthreads();

    // Commit state and write out[:, i+1, :].
    for (int idx = tid; idx < BM * HIDDEN; idx += BLOCK) {
      const int r = idx / HIDDEN, c = idx - r * HIDDEN;
      const float v = xn[idx];
      xb[idx] = v;
      P.out[(size_t)(b0 + r) * ROWSTR + (size_t)(i + 1) * CH + 1 + c] = v;
    }
    const float tn = P.ts[i + 1];
    for (int r = tid; r < BM; r += BLOCK)
      P.out[(size_t)(b0 + r) * ROWSTR + (size_t)(i + 1) * CH] = tn;
    __syncthreads();
  }
}

// fp32 [K,N] row-major -> bf16 WMMA B-fragment layout:
// frag (kt,nt): lane l holds 16 contiguous bf16 = W[kt*32 + (l>>4)*8 + (i&7) + (i>>3)*16][nt*16 + (l&15)],
// zero-padded where k >= K.
__global__ void pack_weights_kernel(const float* __restrict__ W, __bf16* __restrict__ dst,
                                    int K, int N, int KT, int NT) {
  const size_t total = (size_t)KT * NT * 512;
  for (size_t p = (size_t)blockIdx.x * blockDim.x + threadIdx.x; p < total;
       p += (size_t)gridDim.x * blockDim.x) {
    const int    i    = (int)(p & 15);
    const int    lane = (int)((p >> 4) & 31);
    const size_t frag = p >> 9;
    const int    nt   = (int)(frag % NT);
    const int    kt   = (int)(frag / NT);
    const int    k    = kt * 32 + (lane >> 4) * 8 + (i & 7) + (i >> 3) * 16;
    const int    n    = nt * 16 + (lane & 15);
    const float  v    = (k < K) ? W[(size_t)k * N + n] : 0.0f;
    dst[p] = (__bf16)v;
  }
}

extern "C" void kernel_launch(void* const* d_in, const int* in_sizes, int n_in,
                              void* d_out, int out_size, void* d_ws, size_t ws_size,
                              hipStream_t stream) {
  (void)in_sizes; (void)n_in; (void)out_size; (void)ws_size;
  // d_in layout (setup_inputs dict order, pytree-flattened):
  // 0:ts 1:nei 2:initial 3:bm, 4..11: drift (W,b)x4, 12..19: diffusion (W,b)x4, 20:batch_size
  SdeParams P;
  P.ts  = (const float*)d_in[0];
  P.nei = (const float*)d_in[1];
  P.x0  = (const float*)d_in[2];
  P.bm  = (const float*)d_in[3];
  P.out = (float*)d_out;

  static const int Ks[8]  = {129, 512, 512, 512, 129, 512, 512, 512};
  static const int Ns[8]  = {512, 512, 512, 64,  512, 512, 512, 2048};
  static const int KTs[8] = {5, 16, 16, 16, 5, 16, 16, 16};
  static const int NTs[8] = {32, 32, 32, 4, 32, 32, 32, 128};

  __bf16* ws = (__bf16*)d_ws;
  size_t off = 0;
  for (int l = 0; l < 8; ++l) {
    const float* W = (const float*)d_in[4 + 2 * l];
    P.w[l] = ws + off;
    P.b[l] = (const float*)d_in[5 + 2 * l];
    const size_t total = (size_t)KTs[l] * NTs[l] * 512;
    const int blocks = (int)((total + 255) / 256);
    pack_weights_kernel<<<blocks, 256, 0, stream>>>(W, ws + off, Ks[l], Ns[l],
                                                    KTs[l], NTs[l]);
    off += total;
  }

  const size_t shmem = (size_t)BM * KP0 * 2      // z
                     + 2 * (size_t)BM * MLPW * 2 // hA + hB
                     + 2 * (size_t)BM * HIDDEN * 4
                     + (size_t)BM * NOISE * 4;   // ~96.25 KB
  sde_main_kernel<<<BATCH / BM, BLOCK, shmem, stream>>>(P);
}
